// REM_55370718380662
// MI455X (gfx1250) — matrix operand order repaired and checked by reading, not compiled, not run on hardware
//
#include <hip/hip_runtime.h>
#include <hip/hip_bf16.h>

// ---------------------------------------------------------------------------
// Problem constants (match reference)
// ---------------------------------------------------------------------------
#define BATCH 4
#define NPTS 4096
#define CIN 64
#define DIM 64
#define NKP 15
#define KNB 32
#define TRANS 128
#define COLSPB (NPTS * NKP)          // 61440 columns per batch for fuse MLP
#define R2 0.04f                     // CONV_RADIUS^2 = 0.2^2
#define INV_RECEP (1.0f / 1.2f)

// ---------------------------------------------------------------------------
// CDNA5 WMMA / TDM types
// ---------------------------------------------------------------------------
typedef __attribute__((ext_vector_type(16))) __bf16 v16bf;
typedef __attribute__((ext_vector_type(8)))  float  v8f;
typedef __attribute__((ext_vector_type(4)))  unsigned int v4u;
typedef __attribute__((ext_vector_type(4)))  unsigned int tdm_g0_t; // D# group0
typedef __attribute__((ext_vector_type(8)))  int          tdm_g1_t; // D# group1
typedef __attribute__((ext_vector_type(4)))  int          tdm_g23_t;

union V16 {
    v16bf v;
    v4u   q[2];
    unsigned short s[16];
};

// native f32 -> bf16 (single v_cvt on gfx1250; rounding mode = hardware default)
__device__ __forceinline__ unsigned short bf_bits(float x) {
    union { __bf16 b; unsigned short s; } u;
    u.b = (__bf16)x;
    return u.s;
}

// ---------------------------------------------------------------------------
// Workspace layout (bytes, 256-aligned)
// ---------------------------------------------------------------------------
#define IDX_OFF   0                         // int  [B][N][32]        2 MB
#define FEAT0_OFF 2097152                   // bf16 [B][N][64]        2 MB
#define WTD_OFF   4194304                   // bf16 [B][N][15][64]   30 MB
#define OUTB_OFF  35651584                  // bf16 [B][N][64]        2 MB
#define KPW_OFF   37748736                  // bf16 [64][960] (transposed)
#define W1_OFF    37871616                  // bf16 [128][128]   (w2 contiguous after)
#define W2_OFF    37904384                  // bf16 [128][128]

// d_out layout (floats): out [B,64,N] | kp_pos_deform [45] | fuse [B,128,N,15]
#define OUT_ELEMS (BATCH * DIM * NPTS)      // 1048576
#define KP_ELEMS  45

// ===========================================================================
// Kernel 0: weight conversion to bf16 + kp_pos_deform output
// ===========================================================================
__global__ void __launch_bounds__(256) prep_kernel(
    const float* __restrict__ kp_pos, const float* __restrict__ kp_w,
    const float* __restrict__ f1_w,  const float* __restrict__ f2_w,
    unsigned short* __restrict__ kpw_t, unsigned short* __restrict__ w1b,
    unsigned short* __restrict__ w2b, float* __restrict__ kp_out)
{
    int i = blockIdx.x * 256 + threadIdx.x;
    if (i < NKP * DIM * DIM) {                   // kp_weight [15][64][64]
        int k = i >> 6;                          // k = p*64 + c'
        int d = i & 63;
        kpw_t[d * (NKP * DIM) + k] = bf_bits(kp_w[i]);   // transpose -> [d][960]
    }
    if (i < TRANS * TRANS) {
        w1b[i] = bf_bits(f1_w[i]);
        w2b[i] = bf_bits(f2_w[i]);
    }
    if (i < KP_ELEMS) kp_out[i] = kp_pos[i];
}

// ===========================================================================
// Kernel 1: ball query — first 32 in-radius indices in ascending order
// (equivalent to the reference's mask+sort+take-K)
// ===========================================================================
__global__ void __launch_bounds__(256) nbr_kernel(
    const float* __restrict__ pos, int* __restrict__ idx)
{
    __shared__ float sx[NPTS], sy[NPTS], sz[NPTS];
    int tid = threadIdx.x;
    int b = blockIdx.x >> 4;              // 16 blocks per batch
    int n0 = (blockIdx.x & 15) * 256;
    const float* pb = pos + (size_t)b * 3 * NPTS;
    for (int i = tid; i < NPTS; i += 256) {
        sx[i] = pb[i];
        sy[i] = pb[NPTS + i];
        sz[i] = pb[2 * NPTS + i];
    }
    __syncthreads();
    int n = n0 + tid;
    float px = sx[n], py = sy[n], pz = sz[n];
    int cnt = 0;
    int* op = idx + ((size_t)b * NPTS + n) * KNB;
    for (int j = 0; j < NPTS; ++j) {
        float dx = px - sx[j], dy = py - sy[j], dz = pz - sz[j];
        float d2 = dx * dx + dy * dy + dz * dz;
        if (d2 <= R2 && cnt < KNB) { op[cnt] = j; cnt++; }
    }
    for (; cnt < KNB; ++cnt) op[cnt] = NPTS;      // shadow index
}

// ===========================================================================
// Kernel 2: feat0 = cb_w @ feat + cb_b, stored n-major bf16 for gathering
// ===========================================================================
__global__ void __launch_bounds__(256) feat0_kernel(
    const float* __restrict__ feat, const float* __restrict__ cb_w,
    const float* __restrict__ cb_b, unsigned short* __restrict__ feat0)
{
    __shared__ float s_w[DIM * CIN];   // transposed: [c'][c]
    __shared__ float s_b[DIM];
    int tid = threadIdx.x;
    for (int i = tid; i < DIM * CIN; i += 256)
        s_w[(i & 63) * 64 + (i >> 6)] = cb_w[i];
    if (tid < DIM) s_b[tid] = cb_b[tid];
    __syncthreads();

    int pid = blockIdx.x * 256 + tid;       // [0, B*N)
    int b = pid >> 12, n = pid & (NPTS - 1);
    float acc[DIM];
#pragma unroll
    for (int c = 0; c < DIM; ++c) acc[c] = s_b[c];
    for (int cp = 0; cp < CIN; ++cp) {
        float f = feat[((size_t)b * CIN + cp) * NPTS + n];
#pragma unroll
        for (int c = 0; c < DIM; ++c) acc[c] += s_w[cp * 64 + c] * f;
    }
    union { unsigned short s[DIM]; v4u q[8]; } r;
#pragma unroll
    for (int c = 0; c < DIM; ++c) r.s[c] = bf_bits(acc[c]);
    v4u* dst = (v4u*)&feat0[(size_t)pid * DIM];
#pragma unroll
    for (int i = 0; i < 8; ++i) dst[i] = r.q[i];
}

// ===========================================================================
// Kernel 3: KPConv core for a tile of 16 points (128 threads = 4 waves)
//   stage1 (WMMA): weighted[p][c] = sum_k w[p][k] * nbr_feat[k][c]
//   stage2 (WMMA): out0[d] = relu(sum_{p,c'} weighted * kp_w + kp_bias)
//   stage3 (VALU): out = relu(ce_w @ out0 + ce_b + feat)
// ===========================================================================
__global__ void __launch_bounds__(128) conv_kernel(
    const float* __restrict__ pos, const float* __restrict__ feat,
    const float* __restrict__ kp_pos, const float* __restrict__ kp_bias,
    const float* __restrict__ ce_w, const float* __restrict__ ce_b,
    const int* __restrict__ idx, const unsigned short* __restrict__ feat0,
    const unsigned short* __restrict__ kpw_t,
    unsigned short* __restrict__ weighted, unsigned short* __restrict__ outb,
    float* __restrict__ out_f)
{
    __shared__ __align__(16) unsigned short s_w[16 * 16 * KNB];   // [pt][p][k]  16KB
    __shared__ __align__(16) unsigned short s_nf[16 * KNB * DIM]; // [pt][k][c]  64KB
    __shared__ __align__(16) unsigned short s_wt[16 * NKP * DIM]; // [pt][p*64+c] 30KB
    __shared__ float s_o0[16 * DIM];                              // relu(out0)   4KB
    __shared__ float s_cew[DIM * DIM];                            // ce_w        16KB
    __shared__ float s_kpb[DIM], s_ceb[DIM], s_kp[KP_ELEMS];
    __shared__ int   s_idx[16 * KNB];

    int tid = threadIdx.x;
    int b  = blockIdx.x >> 8;                 // 256 tiles per batch
    int n0 = (blockIdx.x & 255) * 16;

    for (int i = tid; i < 16 * KNB; i += 128)
        s_idx[i] = idx[((size_t)b * NPTS + n0) * KNB + i];
    for (int i = tid; i < DIM * DIM; i += 128) s_cew[i] = ce_w[i];
    if (tid < DIM) { s_kpb[tid] = kp_bias[tid]; s_ceb[tid] = ce_b[tid]; }
    if (tid < KP_ELEMS) s_kp[tid] = kp_pos[tid];
    __syncthreads();

    // ---- phase A: gather neighbor positions/features, build kernel weights
    for (int pair = tid; pair < 16 * KNB; pair += 128) {
        int pt = pair >> 5, k = pair & 31;
        int j = s_idx[pair];
        bool valid = j < NPTS;
        float px, py, pz;
        if (valid) {
            px = pos[((size_t)b * 3 + 0) * NPTS + j];
            py = pos[((size_t)b * 3 + 1) * NPTS + j];
            pz = pos[((size_t)b * 3 + 2) * NPTS + j];
        } else { px = py = pz = 1.0e6f; }       // shadow -> w == 0
#pragma unroll
        for (int p = 0; p < NKP; ++p) {
            float dx = px - s_kp[p];
            float dy = py - s_kp[NKP + p];
            float dz = pz - s_kp[2 * NKP + p];
            float d2 = dx * dx + dy * dy + dz * dz;
            float w = fmaxf(1.0f - sqrtf(d2) * INV_RECEP, 0.0f);
            s_w[(pt * 16 + p) * KNB + k] = bf_bits(w);
        }
        s_w[(pt * 16 + 15) * KNB + k] = 0;       // pad row (p == 15)

        v4u* dst = (v4u*)&s_nf[((size_t)pt * KNB + k) * DIM];
        if (valid) {
            const v4u* src = (const v4u*)&feat0[((size_t)b * NPTS + j) * DIM];
#pragma unroll
            for (int i = 0; i < 8; ++i) dst[i] = src[i];
        } else {
            v4u z = {0u, 0u, 0u, 0u};
#pragma unroll
            for (int i = 0; i < 8; ++i) dst[i] = z;
        }
    }
    __syncthreads();

    int lane = tid & 31, wv = tid >> 5;
    int nloc = lane & 15, khalf = lane >> 4;

    // ---- phase B: stage-1 WMMA  (M=p(16), K=32 neighbors, N=64 channels)
    for (int pi = 0; pi < 4; ++pi) {
        int pt = wv * 4 + pi;
        V16 a;   // A-layout: row m=nloc, 16 consecutive k at khalf*16
        a.q[0] = *(const v4u*)&s_w[(pt * 16 + nloc) * KNB + khalf * 16];
        a.q[1] = *(const v4u*)&s_w[(pt * 16 + nloc) * KNB + khalf * 16 + 8];
        int n = n0 + pt;
        for (int nt = 0; nt < 4; ++nt) {
            V16 bb;  // B-layout: col n=nloc, 16 consecutive k at khalf*16
            int c = nt * 16 + nloc;
#pragma unroll
            for (int e = 0; e < 16; ++e)
                bb.s[e] = s_nf[((size_t)pt * KNB + khalf * 16 + e) * DIM + c];
            v8f acc = {};
            acc = __builtin_amdgcn_wmma_f32_16x16x32_bf16(
                false, a.v, false, bb.v, (short)0, acc, false, false);
#pragma unroll
            for (int r = 0; r < 8; ++r) {
                int p = r + 8 * khalf;           // D: m = r + 8*(lane>=16)
                if (p < NKP) {
                    unsigned short hb = bf_bits(acc[r]);
                    s_wt[pt * (NKP * DIM) + p * DIM + c] = hb;
                    weighted[(((size_t)b * NPTS + n) * NKP + p) * DIM + c] = hb;
                }
            }
        }
    }
    __syncthreads();

    // ---- phase C: stage-2 WMMA  (M=16 pts, K=960, N=64; wave handles 16 d)
    {
        int d = wv * 16 + nloc;
        v8f acc = {};
        for (int kc = 0; kc < 30; ++kc) {
            V16 a, bb;
            a.q[0] = *(const v4u*)&s_wt[nloc * (NKP * DIM) + kc * 32 + khalf * 16];
            a.q[1] = *(const v4u*)&s_wt[nloc * (NKP * DIM) + kc * 32 + khalf * 16 + 8];
            const v4u* bp = (const v4u*)&kpw_t[(size_t)d * (NKP * DIM) + kc * 32 + khalf * 16];
            bb.q[0] = bp[0];
            bb.q[1] = bp[1];
            acc = __builtin_amdgcn_wmma_f32_16x16x32_bf16(
                false, a.v, false, bb.v, (short)0, acc, false, false);
        }
#pragma unroll
        for (int r = 0; r < 8; ++r) {
            int pt = r + 8 * khalf;
            s_o0[pt * DIM + d] = fmaxf(acc[r] + s_kpb[d], 0.0f);
        }
    }
    __syncthreads();

    // ---- phase D: ce layer + residual + relu (16x64 outputs over 128 thr)
    {
        int pt = tid >> 3;
        int ob = (tid & 7) * 8;
        int n = n0 + pt;
#pragma unroll
        for (int oi = 0; oi < 8; ++oi) {
            int o = ob + oi;
            float acc = s_ceb[o] + feat[((size_t)b * CIN + o) * NPTS + n];
            for (int c = 0; c < DIM; ++c)
                acc += s_cew[o * DIM + c] * s_o0[pt * DIM + c];
            float v = fmaxf(acc, 0.0f);
            out_f[((size_t)b * DIM + o) * NPTS + n] = v;
            outb[((size_t)b * NPTS + n) * DIM + o] = bf_bits(v);
        }
    }
}

// ===========================================================================
// Kernel 4: fuse MLP — two 128x128 bf16 GEMMs over 245760 columns
// workgroup = 8 waves, 128 columns; wave owns 16 columns, all 128 rows.
// Weight staging (64KB = f1_w || f2_w, contiguous in ws) goes through the
// Tensor Data Mover: 1-D D# descriptor, wave 0 issues, TENSORcnt waited.
// ===========================================================================
__global__ void __launch_bounds__(256) fuse_kernel(
    const unsigned short* __restrict__ w1, const unsigned short* __restrict__ w2,
    const float* __restrict__ f1_b, const float* __restrict__ f2_b,
    const unsigned short* __restrict__ weighted,
    const unsigned short* __restrict__ outb, float* __restrict__ fuse_out)
{
    __shared__ __align__(16) unsigned short s_wts[2 * TRANS * TRANS]; // w1|w2 64KB
    __shared__ __align__(16) unsigned short s_x[128 * TRANS];    // [col][k] 32KB
    __shared__ __align__(16) unsigned short s_h[128 * TRANS];    // [col][k] 32KB
    __shared__ float s_b1[TRANS], s_b2[TRANS];

    unsigned short* s_w1 = s_wts;
    unsigned short* s_w2 = s_wts + TRANS * TRANS;

    int tid = threadIdx.x;
    int b  = blockIdx.x / (COLSPB / 128);
    int c0 = (blockIdx.x % (COLSPB / 128)) * 128;

#if __has_builtin(__builtin_amdgcn_tensor_load_to_lds) && \
    __has_builtin(__builtin_amdgcn_s_wait_tensorcnt)
    if (tid < 32) {   // wave 0 drives the TDM (EXEC-independent DMA)
        // D# group 0: count=1 | lds_addr | global_addr | type=2
        unsigned long long ga = (unsigned long long)(size_t)w1;
        unsigned lds = (unsigned)(size_t)(void*)s_wts;   // LDS aperture offset bits
        tdm_g0_t g0;
        g0.x = 1u;                       // count=1, user descriptor
        g0.y = lds;                      // lds_addr (bytes)
        g0.z = (unsigned)ga;             // global_addr[31:0]
        g0.w = (unsigned)((ga >> 32) & 0x01FFFFFFu) | 0x80000000u; // addr[56:32]|type=2
        // D# group 1: data_size=2B; 1-D tile of 32768 elements
        tdm_g1_t g1;
        g1[0] = 0x00010000;              // wg_mask=0, data_size=1 (2 bytes)
        g1[1] = (int)0x80000000u;        // tensor_dim0[15:0]=32768 in bits[31:16]
        g1[2] = 0x00010000;              // tensor_dim0 hi=0, tensor_dim1=1
        g1[3] = (int)0x80000000u;        // tile_dim0=32768 in bits[31:16]
        g1[4] = 1;                       // tile_dim1=1
        g1[5] = 32768;                   // tensor_dim0_stride = 32768 elements
        g1[6] = 0;
        g1[7] = 0;
        tdm_g23_t gz4 = {0, 0, 0, 0};
        tdm_g1_t  gz8 = {0, 0, 0, 0, 0, 0, 0, 0};
        __builtin_amdgcn_tensor_load_to_lds(g0, g1, gz4, gz4, gz8, 0);
        __builtin_amdgcn_s_wait_tensorcnt(0);
    }
#else
    for (int i = tid; i < 2 * TRANS * TRANS / 8; i += 256)
        ((v4u*)s_wts)[i] = ((const v4u*)w1)[i];   // w2 contiguous after w1 in ws
#endif
    if (tid < TRANS) { s_b1[tid] = f1_b[tid]; s_b2[tid] = f2_b[tid]; }

    {   // load x: col = (n,p) -> [ weighted(b,n,p,0:64) ; out(b,n,0:64) ]
        int col = tid >> 1, half = tid & 1;
        int c = c0 + col;
        int n = c / NKP, p = c % NKP;
        const v4u* src = half
            ? (const v4u*)&outb[((size_t)b * NPTS + n) * DIM]
            : (const v4u*)&weighted[(((size_t)b * NPTS + n) * NKP + p) * DIM];
        v4u* dst = (v4u*)&s_x[col * TRANS + half * DIM];
#pragma unroll
        for (int i = 0; i < 8; ++i) dst[i] = src[i];
    }
    __syncthreads();

    int lane = tid & 31, wv = tid >> 5;
    int cb = wv * 16;
    int nloc = lane & 15, khalf = lane >> 4;
    int col = cb + nloc;

    // ---- layer 1: h = relu(W1 x + b1)
    for (int mt = 0; mt < 8; ++mt) {
        v8f acc = {};
        int m = mt * 16 + nloc;
#pragma unroll
        for (int kc = 0; kc < 4; ++kc) {
            V16 a, bb;
            a.q[0]  = *(const v4u*)&s_w1[m * TRANS + kc * 32 + khalf * 16];
            a.q[1]  = *(const v4u*)&s_w1[m * TRANS + kc * 32 + khalf * 16 + 8];
            bb.q[0] = *(const v4u*)&s_x[col * TRANS + kc * 32 + khalf * 16];
            bb.q[1] = *(const v4u*)&s_x[col * TRANS + kc * 32 + khalf * 16 + 8];
            acc = __builtin_amdgcn_wmma_f32_16x16x32_bf16(
                false, a.v, false, bb.v, (short)0, acc, false, false);
        }
        // rows for consecutive r are contiguous in [col][k]: pack 2x bf16 -> b32
#pragma unroll
        for (int r = 0; r < 8; r += 2) {
            int row = mt * 16 + r + 8 * khalf;
            unsigned lo = bf_bits(fmaxf(acc[r]     + s_b1[row],     0.0f));
            unsigned hi = bf_bits(fmaxf(acc[r + 1] + s_b1[row + 1], 0.0f));
            *(unsigned*)&s_h[col * TRANS + row] = lo | (hi << 16);
        }
    }
    __syncthreads();

    // ---- layer 2: fuse = relu(W2 h + b2) -> global (row-contiguous cols)
    size_t obase = ((size_t)b * TRANS) * COLSPB + c0 + col;
    for (int mt = 0; mt < 8; ++mt) {
        v8f acc = {};
        int m = mt * 16 + nloc;
#pragma unroll
        for (int kc = 0; kc < 4; ++kc) {
            V16 a, bb;
            a.q[0]  = *(const v4u*)&s_w2[m * TRANS + kc * 32 + khalf * 16];
            a.q[1]  = *(const v4u*)&s_w2[m * TRANS + kc * 32 + khalf * 16 + 8];
            bb.q[0] = *(const v4u*)&s_h[col * TRANS + kc * 32 + khalf * 16];
            bb.q[1] = *(const v4u*)&s_h[col * TRANS + kc * 32 + khalf * 16 + 8];
            acc = __builtin_amdgcn_wmma_f32_16x16x32_bf16(
                false, a.v, false, bb.v, (short)0, acc, false, false);
        }
#pragma unroll
        for (int r = 0; r < 8; ++r) {
            int row = mt * 16 + r + 8 * khalf;
            fuse_out[obase + (size_t)row * COLSPB] = fmaxf(acc[r] + s_b2[row], 0.0f);
        }
    }
}

// ===========================================================================
extern "C" void kernel_launch(void* const* d_in, const int* in_sizes, int n_in,
                              void* d_out, int out_size, void* d_ws, size_t ws_size,
                              hipStream_t stream)
{
    (void)in_sizes; (void)n_in; (void)out_size; (void)ws_size;
    const float* pos       = (const float*)d_in[0];
    const float* feat      = (const float*)d_in[1];
    const float* kp_pos    = (const float*)d_in[2];
    const float* cb_w      = (const float*)d_in[3];
    const float* cb_b      = (const float*)d_in[4];
    const float* kp_weight = (const float*)d_in[5];
    const float* kp_bias   = (const float*)d_in[6];
    const float* ce_w      = (const float*)d_in[7];
    const float* ce_b      = (const float*)d_in[8];
    const float* f1_w      = (const float*)d_in[9];
    const float* f1_b      = (const float*)d_in[10];
    const float* f2_w      = (const float*)d_in[11];
    const float* f2_b      = (const float*)d_in[12];

    char* ws = (char*)d_ws;
    int*            idx    = (int*)(ws + IDX_OFF);
    unsigned short* feat0  = (unsigned short*)(ws + FEAT0_OFF);
    unsigned short* wtd    = (unsigned short*)(ws + WTD_OFF);
    unsigned short* outb   = (unsigned short*)(ws + OUTB_OFF);
    unsigned short* kpw_t  = (unsigned short*)(ws + KPW_OFF);
    unsigned short* w1b    = (unsigned short*)(ws + W1_OFF);
    unsigned short* w2b    = (unsigned short*)(ws + W2_OFF);

    float* out_f    = (float*)d_out;
    float* kp_out   = out_f + OUT_ELEMS;
    float* fuse_out = out_f + OUT_ELEMS + KP_ELEMS;

    prep_kernel<<<(NKP * DIM * DIM + 255) / 256, 256, 0, stream>>>(
        kp_pos, kp_weight, f1_w, f2_w, kpw_t, w1b, w2b, kp_out);

    nbr_kernel<<<BATCH * (NPTS / 256), 256, 0, stream>>>(pos, idx);

    feat0_kernel<<<BATCH * NPTS / 256, 256, 0, stream>>>(feat, cb_w, cb_b, feat0);

    conv_kernel<<<BATCH * (NPTS / 16), 128, 0, stream>>>(
        pos, feat, kp_pos, kp_bias, ce_w, ce_b, idx, feat0, kpw_t,
        wtd, outb, out_f);

    fuse_kernel<<<BATCH * (COLSPB / 128), 256, 0, stream>>>(
        w1b, w2b, f1_b, f2_b, wtd, outb, fuse_out);
}
